// _NTXentLoss_89781996356027
// MI455X (gfx1250) — matrix operand order repaired and checked by reading, not compiled, not run on hardware
//
#include <hip/hip_runtime.h>

// ---------------------------------------------------------------------------
// NT-Xent loss, MI455X (gfx1250, wave32, WMMA).
// sim = zn @ zn^T / tau is a 4096x4096x512 GEMM (17.2 GFLOP) -> f16 WMMA,
// fp32 accumulate, fused exp-sum softmax (no 64MB sim materialization).
// Since sim/tau in [-5,5], exp never overflows -> no running max needed.
// A-fragments are held resident in >256 VGPRs (gfx1250 VGPR-MSB), B-fragments
// are double-buffered and streamed from L2 with base+immediate b128 loads.
// ---------------------------------------------------------------------------

typedef __attribute__((ext_vector_type(16))) _Float16 v16h;
typedef __attribute__((ext_vector_type(8)))  _Float16 h8;
typedef __attribute__((ext_vector_type(8)))  float    v8f;

#define DIM     512
#define NROW    4096
#define HALFN   2048
#define INVTAU  5.0f
#define NEG_INF -1.0e9f

// Packed tile strides (in halves): one 16x32 (or 32x16) tile = 512 halves,
// one 16-row (or 16-col) super-tile of 16 k-tiles = 8192 halves.
#define KSTEP   512
#define TSTEP   8192

union Frag { v16h h; uint4 q[2]; };

__device__ __forceinline__ v16h load_frag(const _Float16* __restrict__ p) {
  Frag f;
  f.q[0] = *reinterpret_cast<const uint4*>(p);
  f.q[1] = *reinterpret_cast<const uint4*>(p + 8);
  return f.h;
}

__device__ __forceinline__ v8f wmma_f16(v16h a, v16h b, v8f c) {
  // (neg_a, A, neg_b, B, c_mod, C, reuse_a, reuse_b)
  return __builtin_amdgcn_wmma_f32_16x16x32_f16(false, a, false, b,
                                                (short)0, c, false, false);
}

// ---------------------------------------------------------------------------
// Kernel 1: row-normalize z=[z1;z2] and emit
//   zn  : plain f16 [4096][512]        (for positive-pair dot)
//   Ap  : WMMA A-fragment-ready layout [rt][kt][lane][e]   (16x32 tiles)
//   Bp  : WMMA B-fragment-ready layout [ct][kt][lane][e]   (32x16 tiles)
// A 16-bit layout (ISA 7.12.2):
//   lane<16 : e0..7 -> K=0..7,  e8..15 -> K=16..23   (row = lane&15)
//   lane>=16: e0..7 -> K=8..15, e8..15 -> K=24..31
// B 16-bit layout: same lane/element K order as A, with N = lane&15 in place of the row.
// ---------------------------------------------------------------------------
__global__ __launch_bounds__(128) void nrm_pack_kernel(
    const float* __restrict__ z1, const float* __restrict__ z2,
    _Float16* __restrict__ zn, _Float16* __restrict__ Ap,
    _Float16* __restrict__ Bp) {
  const int row = blockIdx.x;            // 0..4095
  const int t   = threadIdx.x;           // 0..127, 4 floats each
  const float* __restrict__ src =
      (row < HALFN) ? (z1 + (size_t)row * DIM)
                    : (z2 + (size_t)(row - HALFN) * DIM);

  float4 v = reinterpret_cast<const float4*>(src)[t];
  float ss = v.x * v.x + v.y * v.y + v.z * v.z + v.w * v.w;
#pragma unroll
  for (int m = 1; m < 32; m <<= 1) ss += __shfl_xor(ss, m, 32);

  __shared__ float red[4];
  if ((t & 31) == 0) red[t >> 5] = ss;
  __syncthreads();
  const float tot   = red[0] + red[1] + red[2] + red[3];
  const float scale = 1.0f / fmaxf(sqrtf(tot), 1e-8f);

  float e4[4] = {v.x, v.y, v.z, v.w};
  const int rt = row >> 4;     // 16-row tile index
  const int rl = row & 15;     // row within tile
#pragma unroll
  for (int i = 0; i < 4; ++i) {
    const int c = 4 * t + i;                 // column 0..511 (k index)
    const _Float16 h = (_Float16)(e4[i] * scale);
    zn[(size_t)row * DIM + c] = h;

    const int kt = c >> 5;                   // 32-k tile
    const int kl = c & 31;                   // k within tile
    // A-pack
    const int alane = rl | (((kl >> 3) & 1) << 4);
    const int ae    = (kl & 7) | ((kl >> 4) << 3);
    Ap[(size_t)(rt * TSTEP + kt * KSTEP + alane * 16 + ae)] = h;
    // B-pack (this zn row is output-column rt*16+rl)
    Bp[(size_t)(rt * TSTEP + kt * KSTEP + alane * 16 + ae)] = h;
  }
}

// ---------------------------------------------------------------------------
// Kernel 2: fused WMMA GEMM + exp-sum.
// Grid: (128 rowblocks of 32 rows) x (4 column chunks of 1024).
// Block: 128 threads = 4 waves; wave w sweeps 4 column-steps of 64 columns.
// Each wave: 2x4 register-blocked 16x16x32 WMMA tiles (32 rows x 64 cols).
// All 32 A-fragments (32 rows x 512 k) preloaded into registers; B streamed
// with explicit k+1 double buffering so WMMAs overlap outstanding loads.
// Output: psum[row][chunk] partial exp-sums (diagonal masked to exp(-1e9)=0).
// ---------------------------------------------------------------------------
__global__ __launch_bounds__(128) void sim_expsum_kernel(
    const _Float16* __restrict__ Ap, const _Float16* __restrict__ Bp,
    float* __restrict__ psum) {
  const int tid  = threadIdx.x;
  const int w    = tid >> 5;
  const int lane = tid & 31;
  const int rb   = blockIdx.x * 32;        // first row of this block
  const int chunkbase = blockIdx.y * 1024; // first column of this chunk

  // C/D layout: VGPR r -> M = r + 8*(lane>=16); lane&15 -> N within tile.
  const int rbase = rb + ((lane >> 4) << 3);
  const int nloc  = lane & 15;

  // Preload all A fragments: 2 row-tiles x 16 k-tiles (256 VGPRs, MSB-addressed).
  const _Float16* __restrict__ pa = Ap + (size_t)((rb >> 4) * TSTEP + lane * 16);
  v16h af0[16], af1[16];
#pragma unroll
  for (int k = 0; k < 16; ++k) {
    af0[k] = load_frag(pa + k * KSTEP);
    af1[k] = load_frag(pa + k * KSTEP + TSTEP);
  }

  float s0[8], s1[8];
#pragma unroll
  for (int r = 0; r < 8; ++r) { s0[r] = 0.0f; s1[r] = 0.0f; }

#pragma unroll 1
  for (int cs = 0; cs < 4; ++cs) {
    const int colbase = chunkbase + cs * 256 + w * 64;
    // Per-cs B base: fragment (u,k) = pb + u*TSTEP + k*KSTEP.
    const _Float16* __restrict__ pb = Bp + (size_t)((colbase >> 4) * TSTEP + lane * 16);

    v8f a0c0 = {}, a0c1 = {}, a0c2 = {}, a0c3 = {};
    v8f a1c0 = {}, a1c1 = {}, a1c2 = {}, a1c3 = {};

    // Software-pipelined B stream: issue k+1 loads before consuming k.
    v16h b0 = load_frag(pb);
    v16h b1 = load_frag(pb + TSTEP);
    v16h b2 = load_frag(pb + 2 * TSTEP);
    v16h b3 = load_frag(pb + 3 * TSTEP);
#pragma unroll
    for (int k = 0; k < 16; ++k) {
      v16h n0 = b0, n1 = b1, n2 = b2, n3 = b3;
      if (k < 15) {
        n0 = load_frag(pb + (k + 1) * KSTEP);
        n1 = load_frag(pb + (k + 1) * KSTEP + TSTEP);
        n2 = load_frag(pb + (k + 1) * KSTEP + 2 * TSTEP);
        n3 = load_frag(pb + (k + 1) * KSTEP + 3 * TSTEP);
      }
      a0c0 = wmma_f16(af0[k], b0, a0c0);
      a1c0 = wmma_f16(af1[k], b0, a1c0);
      a0c1 = wmma_f16(af0[k], b1, a0c1);
      a1c1 = wmma_f16(af1[k], b1, a1c1);
      a0c2 = wmma_f16(af0[k], b2, a0c2);
      a1c2 = wmma_f16(af1[k], b2, a1c2);
      a0c3 = wmma_f16(af0[k], b3, a0c3);
      a1c3 = wmma_f16(af1[k], b3, a1c3);
      b0 = n0; b1 = n1; b2 = n2; b3 = n3;
    }

#define ACCUM(CT, SS, TT, UU)                                \
    _Pragma("unroll")                                        \
    for (int r = 0; r < 8; ++r) {                            \
      float x = CT[r] * INVTAU;                              \
      const int rg = rbase + 16 * (TT) + r;                  \
      const int cg = colbase + 16 * (UU) + nloc;             \
      x = (rg == cg) ? NEG_INF : x;                          \
      SS[r] += __expf(x);                                    \
    }
    ACCUM(a0c0, s0, 0, 0)
    ACCUM(a0c1, s0, 0, 1)
    ACCUM(a0c2, s0, 0, 2)
    ACCUM(a0c3, s0, 0, 3)
    ACCUM(a1c0, s1, 1, 0)
    ACCUM(a1c1, s1, 1, 1)
    ACCUM(a1c2, s1, 1, 2)
    ACCUM(a1c3, s1, 1, 3)
#undef ACCUM
  }

  // Sum over the 16 columns held by each 16-lane group (fixed order).
#pragma unroll
  for (int m = 1; m < 16; m <<= 1) {
#pragma unroll
    for (int r = 0; r < 8; ++r) {
      s0[r] += __shfl_xor(s0[r], m, 32);
      s1[r] += __shfl_xor(s1[r], m, 32);
    }
  }

  __shared__ float ls[4][32];
  if (nloc == 0) {                          // lanes 0 and 16
    const int rr = (lane >> 4) << 3;        // 0 or 8
#pragma unroll
    for (int r = 0; r < 8; ++r) {
      ls[w][r + rr]      = s0[r];
      ls[w][16 + r + rr] = s1[r];
    }
  }
  __syncthreads();

  if (tid < 32) {
    const float tv = ls[0][tid] + ls[1][tid] + ls[2][tid] + ls[3][tid];
    psum[(size_t)(rb + tid) * 4 + blockIdx.y] = tv;
  }
}

// ---------------------------------------------------------------------------
// Kernel 3: per-row loss_i = log(S_i) - sim[i,pos(i)], reduce to mean.
// pos(i) = i XOR 2048. Positive dot computed from the same f16 data with
// fp32 accumulation (consistent with the WMMA path). Deterministic tree sum.
// ---------------------------------------------------------------------------
__global__ __launch_bounds__(256) void finalize_kernel(
    const _Float16* __restrict__ zn, const float* __restrict__ psum,
    float* __restrict__ out) {
  const int t = threadIdx.x;
  float acc = 0.0f;
  for (int i = t; i < NROW; i += 256) {
    const float S = psum[(size_t)i * 4 + 0] + psum[(size_t)i * 4 + 1] +
                    psum[(size_t)i * 4 + 2] + psum[(size_t)i * 4 + 3];
    const int j = i ^ HALFN;
    const h8* __restrict__ av = reinterpret_cast<const h8*>(zn + (size_t)i * DIM);
    const h8* __restrict__ bv = reinterpret_cast<const h8*>(zn + (size_t)j * DIM);
    float dot = 0.0f;
#pragma unroll 4
    for (int kk = 0; kk < DIM / 8; ++kk) {
      const h8 x = av[kk];
      const h8 y = bv[kk];
#pragma unroll
      for (int e = 0; e < 8; ++e) dot += (float)x[e] * (float)y[e];
    }
    acc += logf(S) - dot * INVTAU;
  }
#pragma unroll
  for (int m = 1; m < 32; m <<= 1) acc += __shfl_xor(acc, m, 32);
  __shared__ float red[8];
  if ((t & 31) == 0) red[t >> 5] = acc;
  __syncthreads();
  if (t == 0) {
    float tv = 0.0f;
#pragma unroll
    for (int wv = 0; wv < 8; ++wv) tv += red[wv];
    out[0] = tv / (float)NROW;
  }
}

// ---------------------------------------------------------------------------
// Workspace layout:
//   [0,   4MB)  zn  f16 [4096][512]
//   [4MB, 8MB)  Ap  f16 (A-fragment packed)
//   [8MB,12MB)  Bp  f16 (B-fragment packed)
//   [12MB, +64KB) psum float [4096][4]
// ---------------------------------------------------------------------------
extern "C" void kernel_launch(void* const* d_in, const int* in_sizes, int n_in,
                              void* d_out, int out_size, void* d_ws, size_t ws_size,
                              hipStream_t stream) {
  (void)in_sizes; (void)n_in; (void)out_size; (void)ws_size;
  const float* z1 = (const float*)d_in[0];
  const float* z2 = (const float*)d_in[1];
  float* out = (float*)d_out;

  char* ws = (char*)d_ws;
  const size_t HALF_BYTES = (size_t)NROW * DIM * sizeof(_Float16); // 4 MB
  _Float16* zn = (_Float16*)(ws);
  _Float16* Ap = (_Float16*)(ws + HALF_BYTES);
  _Float16* Bp = (_Float16*)(ws + 2 * HALF_BYTES);
  float*  psum = (float*)  (ws + 3 * HALF_BYTES);

  nrm_pack_kernel<<<NROW, 128, 0, stream>>>(z1, z2, zn, Ap, Bp);
  sim_expsum_kernel<<<dim3(NROW / 32, 4), 128, 0, stream>>>(Ap, Bp, psum);
  finalize_kernel<<<1, 256, 0, stream>>>(zn, psum, out);
}
